// pixelGRUlayer_16269336117470
// MI455X (gfx1250) — matrix operand order repaired and checked by reading, not compiled, not run on hardware
//
#include <hip/hip_runtime.h>
#include <hip/hip_bf16.h>

typedef __attribute__((ext_vector_type(16))) __bf16 v16bf;
typedef __attribute__((ext_vector_type(8)))  float  v8f;

// ---------- helpers ----------
__device__ __forceinline__ unsigned bf16rne(float f) {
  unsigned u = __float_as_uint(f);
  return (u + 0x7FFFu + ((u >> 16) & 1u)) >> 16;
}
__device__ __forceinline__ unsigned pack2(float lo, float hi) {
  return bf16rne(lo) | (bf16rne(hi) << 16);
}
__device__ __forceinline__ float bf16tof(unsigned short u) {
  return __uint_as_float(((unsigned)u) << 16);
}
// fast sigmoid: v_exp + v_rcp, no IEEE-division expansion
__device__ __forceinline__ float fast_sigmoid(float x) {
  return __builtin_amdgcn_rcpf(1.0f + __expf(-x));
}
// fast tanh: native v_tanh_f32 on CDNA5 if the builtin exists, else branch-free exp form
__device__ __forceinline__ float fast_tanh(float x) {
#if __has_builtin(__builtin_amdgcn_tanhf)
  return __builtin_amdgcn_tanhf(x);
#elif __has_builtin(__builtin_amdgcn_tanh_f32)
  return __builtin_amdgcn_tanh_f32(x);
#else
  float ax = __builtin_fabsf(x);
  float e  = __expf(2.0f * ax);
  float t  = 1.0f - 2.0f * __builtin_amdgcn_rcpf(e + 1.0f);
  return __builtin_copysignf(t, x);
#endif
}

// byte offset of element (m, kk) inside a 1KB A-operand block (16x32 bf16, wave32 layout)
// lanes 0-15 hold K in {0..7}U{16..23}; lanes 16-31 hold K in {8..15}U{24..31}
__device__ __forceinline__ int aOff(int m, int kk) {
  int hi = ((kk >= 8 && kk < 16) || kk >= 24) ? 1 : 0;
  int lane = m + (hi ? 16 : 0);
  int k2 = kk - (hi ? 8 : 0);                   // 0..7 or 16..23
  int g = (k2 < 8) ? (k2 >> 1) : (4 + ((k2 - 16) >> 1));
  return lane * 32 + g * 4 + (k2 & 1) * 2;
}
// byte offset of element (n, kk) inside a 1KB B-operand block (32x16 bf16):
// lanes 0-15 hold K in {0..7}U{16..23}; lanes 16-31 hold K in {8..15}U{24..31}
__device__ __forceinline__ int bOff(int n, int kk) {
  return aOff(n, kk);
}

__device__ __forceinline__ v8f wmma_bf16(v16bf a, v16bf b, v8f c) {
  return __builtin_amdgcn_wmma_f32_16x16x32_bf16(false, a, false, b, (short)0, c, false, false);
}

// ---------- constants ----------
#define BB   64
#define TT   2700      // (32-2)*(32-2)*3
#define HID  128
#define PRE  32
#define VMAX 256
#define GATE (3*HID)   // 384

// ---------- stage 0: zero output (borders stay zero, interior overwritten) ----------
__global__ void k_zero(float4* __restrict__ out) {
  size_t i = (size_t)blockIdx.x * blockDim.x + threadIdx.x;
  float4 z = {0.f, 0.f, 0.f, 0.f};
  out[i] = z;
}

// ---------- stage 1: featurizer; writes pre as bf16 in A-operand swizzled blocks ----------
// preSwz layout: block (t, gb) of 1KB;  gb = batch/16, m = batch%16
__global__ void k_pre(const float* __restrict__ x, const float* __restrict__ Wpre,
                      const float* __restrict__ bpre, unsigned char* __restrict__ preSwz) {
  __shared__ float sW[PRE * 14];
  __shared__ float sB[PRE];
  for (int i = threadIdx.x; i < PRE * 14; i += blockDim.x) sW[i] = Wpre[i];
  if (threadIdx.x < PRE) sB[threadIdx.x] = bpre[threadIdx.x];
  __syncthreads();

  int gid = blockIdx.x * blockDim.x + threadIdx.x;   // 0 .. 64*2700-1
  int b = gid / TT;
  int t = gid % TT;
  int cc = t % 3, pix = t / 3, jj = pix % 30, ii = pix / 30;
  const float* xb = x + (size_t)b * 3 * 32 * 32;

  float inp[14];
#pragma unroll
  for (int ch = 0; ch < 3; ++ch) {
    inp[ch]     = xb[ch * 1024 + ii * 32 + jj];          // tl
    inp[3 + ch] = xb[ch * 1024 + ii * 32 + jj + 1];      // tc
    inp[6 + ch] = xb[ch * 1024 + ii * 32 + jj + 2];      // tr
    inp[9 + ch] = xb[ch * 1024 + (ii + 1) * 32 + jj];    // lf
  }
  inp[12] = (cc > 0) ? xb[0 * 1024 + (ii + 1) * 32 + jj + 1] : -1.0f;
  inp[13] = (cc > 1) ? xb[1 * 1024 + (ii + 1) * 32 + jj + 1] : -1.0f;

  float pre[PRE];
#pragma unroll
  for (int p = 0; p < PRE; ++p) {
    float acc = sB[p];
#pragma unroll
    for (int q = 0; q < 14; ++q) acc += sW[p * 14 + q] * inp[q];
    pre[p] = fast_sigmoid(acc);
  }

  unsigned char* blk = preSwz + (size_t)(t * 4 + (b >> 4)) * 1024;
  int m = b & 15;
#pragma unroll
  for (int p = 0; p < PRE; p += 2)
    *(unsigned*)(blk + aOff(m, p)) = pack2(pre[p], pre[p + 1]);
}

// ---------- stage 2: GRU scan (4 workgroups x 16 batch rows, W resident in LDS) ----------
// Per wave, per step: 2 fused r/z tiles + 1 split n-gate tile => branch-free loop body,
// 15 WMMA per wave per step, perfectly balanced across the 8 waves.
__global__ void __launch_bounds__(256)
k_gru(const unsigned char* __restrict__ preSwz,
      const float* __restrict__ Wih, const float* __restrict__ bih,
      const float* __restrict__ Whh, const float* __restrict__ bhh,
      unsigned short* __restrict__ hsBf) {
  __shared__ unsigned char sBwhh[24 * 4 * 1024];   // 96 KB  (B-swizzled bf16 W_hh)
  __shared__ unsigned char sBwih[24 * 1024];       // 24 KB  (B-swizzled bf16 W_ih)
  __shared__ float sGrz[16][256];                  // fused r|z pre-activations
  __shared__ float sXn[16][128];                   // x contribution of n-gate
  __shared__ float sHn[16][128];                   // h contribution of n-gate
  __shared__ unsigned char sAh[4 * 1024];          // A-swizzled bf16 h state
  __shared__ float sBih[GATE];
  __shared__ float sBhh[GATE];

  const int tid = threadIdx.x;
  const int gb  = blockIdx.x;        // batch tile 0..3

  // --- swizzle W_hh (384x128) into B-operand layout ---
  for (int idx = tid; idx < GATE * 64; idx += 256) {
    int n = idx / 64, kp = (idx % 64) * 2;
    unsigned w = pack2(Whh[n * HID + kp], Whh[n * HID + kp + 1]);
    *(unsigned*)(sBwhh + ((n >> 4) * 4 + (kp >> 5)) * 1024 + bOff(n & 15, kp & 31)) = w;
  }
  // --- swizzle W_ih (384x32) ---
  for (int idx = tid; idx < GATE * 16; idx += 256) {
    int n = idx / 16, kp = (idx % 16) * 2;
    unsigned w = pack2(Wih[n * PRE + kp], Wih[n * PRE + kp + 1]);
    *(unsigned*)(sBwih + (n >> 4) * 1024 + bOff(n & 15, kp)) = w;
  }
  for (int i = tid; i < GATE; i += 256) { sBih[i] = bih[i]; sBhh[i] = bhh[i]; }
  for (int i = tid; i < 1024; i += 256) ((unsigned*)sAh)[i] = 0u;   // h0 = 0
  __syncthreads();

  const int wave  = tid >> 5;
  const int lane  = tid & 31;
  const int laneN = lane & 15;
  const int mBase = (lane >> 4) << 3;

  // hoist B operands + biases into registers (constant over the whole scan)
  // bx/bh[0..1]: fused tiles 2w, 2w+1   |   bx/bh[2]: split tile 16+w
  v16bf bx[3], bh[3][4];
  float biasS[2], biasXn, biasHn;
#pragma unroll
  for (int j = 0; j < 2; ++j) {
    int tile = wave * 2 + j;
    bx[j] = *(const v16bf*)(sBwih + tile * 1024 + lane * 32);
#pragma unroll
    for (int c = 0; c < 4; ++c)
      bh[j][c] = *(const v16bf*)(sBwhh + (tile * 4 + c) * 1024 + lane * 32);
    int n = tile * 16 + laneN;
    biasS[j] = sBih[n] + sBhh[n];
  }
  {
    int tile = 16 + wave;
    bx[2] = *(const v16bf*)(sBwih + tile * 1024 + lane * 32);
#pragma unroll
    for (int c = 0; c < 4; ++c)
      bh[2][c] = *(const v16bf*)(sBwhh + (tile * 4 + c) * 1024 + lane * 32);
    int n = tile * 16 + laneN;
    biasXn = sBih[n];
    biasHn = sBhh[n];
  }

  const int gm  = tid >> 4;           // gate-phase row
  const int gn0 = (tid & 15) * 8;     // gate-phase column base (multiple of 8)
  const int ahBase = (gn0 >> 5) * 1024 + aOff(gm, gn0 & 31);  // 16B-aligned run of 8 K

  // software pipeline the A_pre global load across the scan
  v16bf ap = *(const v16bf*)(preSwz + (size_t)gb * 1024 + lane * 32);

  for (int t = 0; t < TT; ++t) {
    int tn = (t + 1 < TT) ? t + 1 : t;
    v16bf apNext = *(const v16bf*)(preSwz + (size_t)(tn * 4 + gb) * 1024 + lane * 32);

    v16bf ah[4];
#pragma unroll
    for (int c = 0; c < 4; ++c)
      ah[c] = *(const v16bf*)(sAh + c * 1024 + lane * 32);

    // ---- 2 fused r/z tiles ----
#pragma unroll
    for (int j = 0; j < 2; ++j) {
      v8f acc;
#pragma unroll
      for (int v = 0; v < 8; ++v) acc[v] = biasS[j];
      acc = wmma_bf16(ap, bx[j], acc);
#pragma unroll
      for (int c = 0; c < 4; ++c) acc = wmma_bf16(ah[c], bh[j][c], acc);
      int n = (wave * 2 + j) * 16 + laneN;
#pragma unroll
      for (int v = 0; v < 8; ++v) sGrz[mBase + v][n] = acc[v];
    }
    // ---- 1 split n-gate tile (x and h contributions kept separate) ----
    {
      v8f ax, ahh;
#pragma unroll
      for (int v = 0; v < 8; ++v) { ax[v] = biasXn; ahh[v] = biasHn; }
      ax = wmma_bf16(ap, bx[2], ax);
#pragma unroll
      for (int c = 0; c < 4; ++c) ahh = wmma_bf16(ah[c], bh[2][c], ahh);
      int n = wave * 16 + laneN;
#pragma unroll
      for (int v = 0; v < 8; ++v) { sXn[mBase + v][n] = ax[v]; sHn[mBase + v][n] = ahh[v]; }
    }
    __syncthreads();

    // ---- gates: each thread owns 8 consecutive state elements -> vector LDS ops ----
    float gr[8], gz[8], xn[8], hn[8];
    {
      const float4* p;
      p = (const float4*)&sGrz[gm][gn0];        *(float4*)&gr[0] = p[0]; *(float4*)&gr[4] = p[1];
      p = (const float4*)&sGrz[gm][gn0 + 128];  *(float4*)&gz[0] = p[0]; *(float4*)&gz[4] = p[1];
      p = (const float4*)&sXn[gm][gn0];         *(float4*)&xn[0] = p[0]; *(float4*)&xn[4] = p[1];
      p = (const float4*)&sHn[gm][gn0];         *(float4*)&hn[0] = p[0]; *(float4*)&hn[4] = p[1];
    }
    uint4 hpv = *(const uint4*)(sAh + ahBase);   // 8 bf16 h_prev in one ds_load_b128
    unsigned hpw[4]; *(uint4*)hpw = hpv;

    unsigned outw[4];
#pragma unroll
    for (int j = 0; j < 4; ++j) {
      float h2[2];
#pragma unroll
      for (int s = 0; s < 2; ++s) {
        int e = j * 2 + s;
        float r  = fast_sigmoid(gr[e]);
        float z  = fast_sigmoid(gz[e]);
        float nv = fast_tanh(xn[e] + r * hn[e]);
        float hp = bf16tof((unsigned short)((s == 0) ? (hpw[j] & 0xFFFFu) : (hpw[j] >> 16)));
        h2[s] = (1.0f - z) * nv + z * hp;
      }
      outw[j] = pack2(h2[0], h2[1]);
    }
    uint4 ov = *(const uint4*)outw;
    *(uint4*)(sAh + ahBase) = ov;                                         // feed next step
    unsigned short* hsRow = hsBf + ((size_t)(gb * 16 + gm) * TT + t) * HID;
    *(uint4*)(hsRow + gn0) = ov;                                          // one b128 store
    __syncthreads();

    ap = apNext;
  }
}

// ---------- stage 3: out = hs @ W_post^T + b_post, scattered into padded output ----------
__global__ void __launch_bounds__(256)
k_post(const unsigned short* __restrict__ hsBf, const float* __restrict__ Wpost,
       const float* __restrict__ bpost, float* __restrict__ out) {
  __shared__ unsigned char sB[16 * 4 * 1024];   // 64 KB  (B-swizzled bf16 W_post)
  __shared__ unsigned char sA[4 * 1024];        // A-swizzled bf16 hs row-tile
  __shared__ float sBias[VMAX];

  const int tid = threadIdx.x;
  for (int idx = tid; idx < VMAX * 64; idx += 256) {
    int n = idx / 64, kp = (idx % 64) * 2;
    unsigned w = pack2(Wpost[n * HID + kp], Wpost[n * HID + kp + 1]);
    *(unsigned*)(sB + ((n >> 4) * 4 + (kp >> 5)) * 1024 + bOff(n & 15, kp & 31)) = w;
  }
  for (int i = tid; i < VMAX; i += 256) sBias[i] = bpost[i];
  __syncthreads();

  const int wave  = tid >> 5;
  const int lane  = tid & 31;
  const int laneN = lane & 15;
  const int mBase = (lane >> 4) << 3;

  // hoist B operands into registers (2 N-tiles per wave)
  v16bf breg[2][4];
  float biasR[2];
#pragma unroll
  for (int h = 0; h < 2; ++h) {
    int tile = wave * 2 + h;
#pragma unroll
    for (int c = 0; c < 4; ++c)
      breg[h][c] = *(const v16bf*)(sB + (tile * 4 + c) * 1024 + lane * 32);
    biasR[h] = sBias[tile * 16 + laneN];
  }

  const int am = tid >> 4, ak0 = (tid & 15) * 8;
  const int aBase = (ak0 >> 5) * 1024 + aOff(am, ak0 & 31);  // contiguous 16B run

  for (int it = 0; it < 16; ++it) {
    int rt = blockIdx.x * 16 + it;               // row tile over flattened (b,t)
    // stage hs rows into A-swizzled LDS (one b128 load + one b128 store per thread)
    const uint4* src = (const uint4*)(hsBf + (size_t)(rt * 16 + am) * HID + ak0);
    *(uint4*)(sA + aBase) = *src;
    __syncthreads();

    v16bf a[4];
#pragma unroll
    for (int c = 0; c < 4; ++c)
      a[c] = *(const v16bf*)(sA + c * 1024 + lane * 32);

#pragma unroll
    for (int h = 0; h < 2; ++h) {
      int n = (wave * 2 + h) * 16 + laneN;
      v8f acc;
#pragma unroll
      for (int v = 0; v < 8; ++v) acc[v] = biasR[h];
#pragma unroll
      for (int c = 0; c < 4; ++c) acc = wmma_bf16(a[c], breg[h][c], acc);
#pragma unroll
      for (int v = 0; v < 8; ++v) {
        int row = rt * 16 + mBase + v;
        int b = row / TT, t = row % TT;
        int cc = t % 3, pix = t / 3, jj = pix % 30, ii = pix / 30;
        size_t idx = (((size_t)(b * 3 + cc) * 32 + ii + 1) * 32 + jj + 1) * 256 + n;
        out[idx] = acc[v];
      }
    }
    __syncthreads();
  }
}

// ---------- launch ----------
extern "C" void kernel_launch(void* const* d_in, const int* in_sizes, int n_in,
                              void* d_out, int out_size, void* d_ws, size_t ws_size,
                              hipStream_t stream) {
  const float* x     = (const float*)d_in[0];
  const float* Wpre  = (const float*)d_in[1];
  const float* bpre  = (const float*)d_in[2];
  const float* Wih   = (const float*)d_in[3];
  const float* bihp  = (const float*)d_in[4];
  const float* Whh   = (const float*)d_in[5];
  const float* bhhp  = (const float*)d_in[6];
  const float* Wpost = (const float*)d_in[7];
  const float* bpost = (const float*)d_in[8];

  unsigned char*  preSwz = (unsigned char*)d_ws;                       // 2700*4*1024 B
  unsigned short* hsBf   = (unsigned short*)((char*)d_ws + (size_t)TT * 4 * 1024);

  k_zero<<<(BB * 3 * 32 * 32 * VMAX) / (256 * 4), 256, 0, stream>>>((float4*)d_out);
  k_pre <<<(BB * TT) / 256, 256, 0, stream>>>(x, Wpre, bpre, preSwz);
  k_gru <<<4, 256, 0, stream>>>(preSwz, Wih, bihp, Whh, bhhp, hsBf);
  k_post<<<(BB * TT) / (16 * 16), 256, 0, stream>>>(hsBf, Wpost, bpost, (float*)d_out);
}
